// TI_TransformerNoEdges_59365037966016
// MI455X (gfx1250) — matrix operand order repaired and checked by reading, not compiled, not run on hardware
//
#include <hip/hip_runtime.h>

#define NN   4096
#define EE   131072
#define EP   (EE + NN)     /* edges + self loops = 135168 */
#define FF   128
#define EMB  128
#define HH   8
#define DC   64
#define DD   512
#define FFH  2048
#define GG   8
#define OUTD 64

typedef __attribute__((ext_vector_type(16))) __bf16 v16bf;
typedef __attribute__((ext_vector_type(8)))  float  v8f;

union Frag16 { uint4 u[2]; v16bf v; };

__device__ __forceinline__ unsigned short f2b(float f) {
    unsigned u = __float_as_uint(f);
    unsigned r = u + 0x7fffu + ((u >> 16) & 1u);
    return (unsigned short)(r >> 16);
}
__device__ __forceinline__ unsigned f2ord(float f) {
    unsigned u = __float_as_uint(f);
    return (u & 0x80000000u) ? ~u : (u | 0x80000000u);
}
__device__ __forceinline__ float ord2f(unsigned o) {
    unsigned u = (o & 0x80000000u) ? (o & 0x7fffffffu) : ~o;
    return __uint_as_float(u);
}

/* ---------------- utility kernels ---------------- */
__global__ void k_fill_f32(float* p, float v, int n) {
    int g = blockIdx.x * 256 + threadIdx.x; if (g < n) p[g] = v;
}
__global__ void k_fill_u32(unsigned* p, unsigned v, int n) {
    int g = blockIdx.x * 256 + threadIdx.x; if (g < n) p[g] = v;
}
__global__ void k_copy_u16(const unsigned short* __restrict__ a, unsigned short* __restrict__ b, int n) {
    int g = blockIdx.x * 256 + threadIdx.x; if (g < n) b[g] = a[g];
}
__global__ void k_cvt_bf16(const float* __restrict__ a, unsigned short* __restrict__ b, int n) {
    int g = blockIdx.x * 256 + threadIdx.x; if (g < n) b[g] = f2b(a[g]);
}
/* pack f32 weight [K x Nc] row-major into WMMA B-fragment order:
   tile (k0/32, n0/16): 512 elems; elem (lane,j): B[k0 + (j>>3)*16 + (lane>>4)*8 + (j&7)][n0 + (lane&15)] */
__global__ void k_pack_b(const float* __restrict__ B, unsigned short* __restrict__ P, int K, int Nc) {
    int gid = blockIdx.x * 256 + threadIdx.x;
    if (gid >= K * Nc) return;
    int tile = gid >> 9, r = gid & 511;
    int lane = r >> 4, j = r & 15;
    int ntiles = Nc >> 4;
    int k0 = (tile / ntiles) << 5;
    int n0 = (tile % ntiles) << 4;
    int k = k0 + ((j >> 3) << 4) + ((lane >> 4) << 3) + (j & 7);
    int n = n0 + (lane & 15);
    P[gid] = f2b(B[(size_t)k * Nc + n]);
}

/* ---------------- bf16 WMMA GEMM: C = act(A@B + bias) ----------------
   block = 256 thr = 8 waves (4 along M x 2 along N); block tile 128x128;
   wave tile 32x64 (2 A-frags x 4 B-frags -> 8 WMMA per 32-deep k-step) */
__global__ __launch_bounds__(256)
void k_gemm_bf16(const unsigned short* __restrict__ A,   // M x K bf16 row-major
                 const unsigned short* __restrict__ Bp,  // packed B
                 const float* __restrict__ bias,
                 float* __restrict__ Cf, unsigned short* __restrict__ Cb,
                 int M, int K, int Nc, int act)
{
    int lane = threadIdx.x & 31, wid = threadIdx.x >> 5;
    int wm = wid & 3, wn = wid >> 2;
    int row0 = blockIdx.y * 128 + wm * 32;
    int col0 = blockIdx.x * 128 + wn * 64;
    int l15 = lane & 15, hi = lane >> 4;
    v8f acc[2][4] = {};
    const unsigned short* Ab0 = A + (size_t)(row0 + l15) * K + hi * 8;
    const unsigned short* Ab1 = Ab0 + (size_t)16 * K;
    int ntiles = Nc >> 4;
    size_t kstride = (size_t)ntiles * 512;
    const unsigned short* bt = Bp + (size_t)(col0 >> 4) * 512 + lane * 16;
    for (int k0 = 0; k0 < K; k0 += 32) {
        Frag16 a0, a1;
        a0.u[0] = *(const uint4*)(Ab0 + k0); a0.u[1] = *(const uint4*)(Ab0 + k0 + 16);
        a1.u[0] = *(const uint4*)(Ab1 + k0); a1.u[1] = *(const uint4*)(Ab1 + k0 + 16);
        __builtin_prefetch(bt + kstride, 0, 1);   /* next k-tile of packed B */
        #pragma unroll
        for (int t = 0; t < 4; ++t) {
            Frag16 b;
            b.u[0] = *(const uint4*)(bt + t * 512);
            b.u[1] = *(const uint4*)(bt + t * 512 + 8);
            acc[0][t] = __builtin_amdgcn_wmma_f32_16x16x32_bf16(false, a0.v, false, b.v, (short)0, acc[0][t], false, false);
            acc[1][t] = __builtin_amdgcn_wmma_f32_16x16x32_bf16(false, a1.v, false, b.v, (short)0, acc[1][t], false, false);
        }
        bt += kstride;
    }
    #pragma unroll
    for (int t = 0; t < 4; ++t) {
        int col = col0 + t * 16 + l15;
        float bv = bias ? bias[col] : 0.f;
        #pragma unroll
        for (int mi = 0; mi < 2; ++mi) {
            #pragma unroll
            for (int r = 0; r < 8; ++r) {
                int row = row0 + mi * 16 + r + hi * 8;
                float xv = acc[mi][t][r] + bv;
                if (act == 1) xv = fmaxf(xv, 0.f);
                else if (act == 2) xv = xv >= 0.f ? xv : 0.01f * xv;
                size_t ba = (size_t)row * Nc + col;
                if (Cf) Cf[ba] = xv;
                if (Cb) Cb[ba] = f2b(xv);
            }
        }
    }
}

/* ---------------- flash attention (per wave: one head, 16 query rows) ---------------- */
__global__ __launch_bounds__(256)
void k_flash(const unsigned short* __restrict__ Q, const unsigned short* __restrict__ Kb,
             const unsigned short* __restrict__ VT,  // [D][N] : VT[c*N + n]
             float* __restrict__ Of, unsigned short* __restrict__ Ob)
{
    __shared__ unsigned short pb[8][16][32];
    int lane = threadIdx.x & 31, wid = threadIdx.x >> 5;
    int w = blockIdx.x * 8 + wid;
    int head = w >> 8, qb = w & 255;
    int l15 = lane & 15, hi = lane >> 4;
    Frag16 aq0, aq1;
    const unsigned short* qrow = Q + (size_t)(qb * 16 + l15) * DD + head * DC + hi * 16;
    aq0.u[0] = *(const uint4*)(qrow);      aq0.u[1] = *(const uint4*)(qrow + 8);
    aq1.u[0] = *(const uint4*)(qrow + 32); aq1.u[1] = *(const uint4*)(qrow + 40);
    v8f o0 = {}, o1 = {}, o2 = {}, o3 = {};
    float mrow[8], lrow[8];
    #pragma unroll
    for (int r = 0; r < 8; ++r) { mrow[r] = -3.0e38f; lrow[r] = 0.f; }
    const float sc = 0.125f; // 1/sqrt(64)
    for (int kb = 0; kb < NN; kb += 32) {
        v8f s0 = {}, s1 = {};
        {
            const unsigned short* kbase = Kb + (size_t)(kb + l15) * DD + head * DC + hi * 16;
            const unsigned short* kbase2 = kbase + (size_t)16 * DD;
            Frag16 b00, b10, b01, b11;
            b00.u[0] = *(const uint4*)(kbase);       b00.u[1] = *(const uint4*)(kbase + 8);
            b10.u[0] = *(const uint4*)(kbase + 32);  b10.u[1] = *(const uint4*)(kbase + 40);
            b01.u[0] = *(const uint4*)(kbase2);      b01.u[1] = *(const uint4*)(kbase2 + 8);
            b11.u[0] = *(const uint4*)(kbase2 + 32); b11.u[1] = *(const uint4*)(kbase2 + 40);
            s0 = __builtin_amdgcn_wmma_f32_16x16x32_bf16(false, aq0.v, false, b00.v, (short)0, s0, false, false);
            s0 = __builtin_amdgcn_wmma_f32_16x16x32_bf16(false, aq1.v, false, b10.v, (short)0, s0, false, false);
            s1 = __builtin_amdgcn_wmma_f32_16x16x32_bf16(false, aq0.v, false, b01.v, (short)0, s1, false, false);
            s1 = __builtin_amdgcn_wmma_f32_16x16x32_bf16(false, aq1.v, false, b11.v, (short)0, s1, false, false);
        }
        #pragma unroll
        for (int r = 0; r < 8; ++r) { s0[r] *= sc; s1[r] *= sc; }
        #pragma unroll
        for (int r = 0; r < 8; ++r) {
            float v = fmaxf(s0[r], s1[r]);
            v = fmaxf(v, __shfl_xor(v, 1, 16));
            v = fmaxf(v, __shfl_xor(v, 2, 16));
            v = fmaxf(v, __shfl_xor(v, 4, 16));
            v = fmaxf(v, __shfl_xor(v, 8, 16));
            float mnew = fmaxf(mrow[r], v);
            float scl = expf(mrow[r] - mnew);
            float p0 = expf(s0[r] - mnew), p1 = expf(s1[r] - mnew);
            s0[r] = p0; s1[r] = p1;
            float rs = p0 + p1;
            rs += __shfl_xor(rs, 1, 16); rs += __shfl_xor(rs, 2, 16);
            rs += __shfl_xor(rs, 4, 16); rs += __shfl_xor(rs, 8, 16);
            lrow[r] = lrow[r] * scl + rs;
            mrow[r] = mnew;
            o0[r] *= scl; o1[r] *= scl; o2[r] *= scl; o3[r] *= scl;
        }
        /* D-layout -> A-layout transpose of P through LDS (same wave) */
        #pragma unroll
        for (int r = 0; r < 8; ++r) {
            pb[wid][r + hi * 8][l15]      = f2b(s0[r]);
            pb[wid][r + hi * 8][16 + l15] = f2b(s1[r]);
        }
        __asm volatile("s_wait_dscnt 0x0" ::: "memory");
        Frag16 pf;
        pf.u[0] = *(const uint4*)&pb[wid][l15][hi * 16];
        pf.u[1] = *(const uint4*)&pb[wid][l15][hi * 16 + 8];
        const unsigned short* vbase = VT + (size_t)(head * DC) * NN + kb + hi * 16;
        Frag16 bv;
        bv.u[0] = *(const uint4*)(vbase + (size_t)(0 * 16 + l15) * NN);
        bv.u[1] = *(const uint4*)(vbase + (size_t)(0 * 16 + l15) * NN + 8);
        o0 = __builtin_amdgcn_wmma_f32_16x16x32_bf16(false, pf.v, false, bv.v, (short)0, o0, false, false);
        bv.u[0] = *(const uint4*)(vbase + (size_t)(1 * 16 + l15) * NN);
        bv.u[1] = *(const uint4*)(vbase + (size_t)(1 * 16 + l15) * NN + 8);
        o1 = __builtin_amdgcn_wmma_f32_16x16x32_bf16(false, pf.v, false, bv.v, (short)0, o1, false, false);
        bv.u[0] = *(const uint4*)(vbase + (size_t)(2 * 16 + l15) * NN);
        bv.u[1] = *(const uint4*)(vbase + (size_t)(2 * 16 + l15) * NN + 8);
        o2 = __builtin_amdgcn_wmma_f32_16x16x32_bf16(false, pf.v, false, bv.v, (short)0, o2, false, false);
        bv.u[0] = *(const uint4*)(vbase + (size_t)(3 * 16 + l15) * NN);
        bv.u[1] = *(const uint4*)(vbase + (size_t)(3 * 16 + l15) * NN + 8);
        o3 = __builtin_amdgcn_wmma_f32_16x16x32_bf16(false, pf.v, false, bv.v, (short)0, o3, false, false);
    }
    #pragma unroll
    for (int r = 0; r < 8; ++r) {
        float inv = 1.0f / lrow[r];
        size_t base = (size_t)(qb * 16 + r + hi * 8) * DD + head * DC + l15;
        float v0 = o0[r] * inv, v1 = o1[r] * inv, v2 = o2[r] * inv, v3 = o3[r] * inv;
        Of[base] = v0; Of[base + 16] = v1; Of[base + 32] = v2; Of[base + 48] = v3;
        Ob[base] = f2b(v0); Ob[base + 16] = f2b(v1); Ob[base + 32] = f2b(v2); Ob[base + 48] = f2b(v3);
    }
}

/* ---------------- GAT helpers ---------------- */
__global__ void k_attn_coef(const float* __restrict__ hw, const float* __restrict__ as_,
                            const float* __restrict__ ad_, float* __restrict__ os, float* __restrict__ od)
{
    int gid = blockIdx.x * 256 + threadIdx.x;
    if (gid >= NN * HH) return;
    int n = gid >> 3, hh = gid & 7;
    const float* hp = hw + (size_t)n * DD + hh * DC;
    const float* ap = as_ + hh * DC;
    const float* dp = ad_ + hh * DC;
    float s1 = 0.f, s2 = 0.f;
    for (int i = 0; i < DC; ++i) { float t = hp[i]; s1 += t * ap[i]; s2 += t * dp[i]; }
    os[gid] = s1; od[gid] = s2;
}
__device__ __forceinline__ void edge_sd(const int* ei, int e, int& s, int& d) {
    if (e < EE) { s = ei[e]; d = ei[EE + e]; } else { s = e - EE; d = e - EE; }
}
__global__ void k_edge_max(const int* __restrict__ ei, const float* __restrict__ as_,
                           const float* __restrict__ ad_, float* __restrict__ el, unsigned* __restrict__ mm)
{
    int gid = blockIdx.x * 256 + threadIdx.x;
    if (gid >= EP * HH) return;
    int e = gid >> 3, hh = gid & 7, s, d;
    edge_sd(ei, e, s, d);
    float ev = as_[s * HH + hh] + ad_[d * HH + hh];
    ev = ev >= 0.f ? ev : 0.2f * ev;
    el[gid] = ev;
    atomicMax(mm + d * HH + hh, f2ord(ev));
}
__global__ void k_edge_exp(const int* __restrict__ ei, float* __restrict__ el,
                           const unsigned* __restrict__ mm, float* __restrict__ den)
{
    int gid = blockIdx.x * 256 + threadIdx.x;
    if (gid >= EP * HH) return;
    int e = gid >> 3, hh = gid & 7, s, d;
    edge_sd(ei, e, s, d);
    float ex = expf(el[gid] - ord2f(mm[d * HH + hh]));
    el[gid] = ex;
    atomicAdd(den + d * HH + hh, ex);
}
__global__ void k_edge_alpha(const int* __restrict__ ei, float* __restrict__ el, const float* __restrict__ den)
{
    int gid = blockIdx.x * 256 + threadIdx.x;
    if (gid >= EP * HH) return;
    int e = gid >> 3, hh = gid & 7, s, d;
    edge_sd(ei, e, s, d);
    el[gid] = el[gid] / (den[d * HH + hh] + 1e-16f);
}
__global__ void k_scatter(const int* __restrict__ ei, const float* __restrict__ al,
                          const float* __restrict__ hw, float* __restrict__ out)
{
    int gid = blockIdx.x * 256 + threadIdx.x;
    if (gid >= EP * DD) return;
    int e = gid >> 9, dd = gid & 511, s, d;
    edge_sd(ei, e, s, d);
    float a = al[e * HH + (dd >> 6)];
    atomicAdd(out + (size_t)d * DD + dd, a * hw[(size_t)s * DD + dd]);
}
__global__ void k_bias_leaky(const float* __restrict__ t2, const float* __restrict__ bb,
                             float* __restrict__ hO, unsigned short* __restrict__ hbO)
{
    int gid = blockIdx.x * 256 + threadIdx.x;
    if (gid >= NN * DD) return;
    float v = t2[gid] + bb[gid & 511];
    v = v >= 0.f ? v : 0.01f * v;
    hO[gid] = v; hbO[gid] = f2b(v);
}

/* ---------------- layernorm with residual (wave per row) ---------------- */
__global__ __launch_bounds__(256)
void k_ln_res(const float* __restrict__ X, const float* __restrict__ Dl,
              const float* __restrict__ S, const float* __restrict__ B,
              float* __restrict__ Yf, unsigned short* __restrict__ Yb)
{
    int lane = threadIdx.x & 31, wid = threadIdx.x >> 5;
    int row = blockIdx.x * 8 + wid;
    const float* x = X + (size_t)row * DD;
    const float* dl = Dl + (size_t)row * DD;
    float v[16], s = 0.f, sq = 0.f;
    #pragma unroll
    for (int i = 0; i < 16; ++i) {
        int c = lane + i * 32;
        float t = x[c] + dl[c];
        v[i] = t; s += t; sq += t * t;
    }
    for (int off = 1; off < 32; off <<= 1) { s += __shfl_xor(s, off); sq += __shfl_xor(sq, off); }
    float mean = s * (1.f / DD);
    float var = sq * (1.f / DD) - mean * mean;
    float inv = rsqrtf(var + 1e-5f);
    #pragma unroll
    for (int i = 0; i < 16; ++i) {
        int c = lane + i * 32;
        float y = (v[i] - mean) * inv * S[c] + B[c];
        Yf[(size_t)row * DD + c] = y; Yb[(size_t)row * DD + c] = f2b(y);
    }
}

__global__ void k_transpose_u16(const unsigned short* __restrict__ Vi, unsigned short* __restrict__ Vo) {
    int gid = blockIdx.x * 256 + threadIdx.x;
    if (gid >= NN * DD) return;
    int n = gid >> 9, c = gid & 511;
    Vo[(size_t)c * NN + n] = Vi[gid];
}

/* ---------------- gate / pooling ---------------- */
__global__ __launch_bounds__(256)
void k_rowdot(const float* __restrict__ X, const float* __restrict__ Wv,
              const float* __restrict__ bsc, float* __restrict__ Out, int Kd)
{
    int lane = threadIdx.x & 31, wid = threadIdx.x >> 5;
    int row = blockIdx.x * 8 + wid;
    const float* x = X + (size_t)row * Kd;
    float s = 0.f;
    for (int c = lane; c < Kd; c += 32) s += x[c] * Wv[c];
    for (int off = 1; off < 32; off <<= 1) s += __shfl_xor(s, off);
    if (lane == 0) Out[row] = s + bsc[0];
}
__global__ void k_gmax(const int* __restrict__ batch, const float* __restrict__ g, unsigned* __restrict__ gm) {
    int n = blockIdx.x * 256 + threadIdx.x;
    if (n >= NN) return;
    atomicMax(gm + batch[n], f2ord(g[n]));
}
__global__ void k_gexp(const int* __restrict__ batch, const float* __restrict__ g,
                       const unsigned* __restrict__ gm, float* __restrict__ ge, float* __restrict__ gs) {
    int n = blockIdx.x * 256 + threadIdx.x;
    if (n >= NN) return;
    float e = expf(g[n] - ord2f(gm[batch[n]]));
    ge[n] = e;
    atomicAdd(gs + batch[n], e);
}
__global__ void k_pool(const int* __restrict__ batch, const float* __restrict__ ge,
                       const float* __restrict__ gs, const float* __restrict__ h, float* __restrict__ pooled) {
    int gid = blockIdx.x * 256 + threadIdx.x;
    if (gid >= NN * DD) return;
    int n = gid >> 9, dd = gid & 511;
    int b = batch[n];
    float w = ge[n] / (gs[b] + 1e-16f);
    atomicAdd(pooled + (size_t)b * DD + dd, w * h[gid]);
}
__global__ void k_fc(const float* __restrict__ pooled, const float* __restrict__ fcw,
                     const float* __restrict__ fcb, float* __restrict__ out) {
    int tid = blockIdx.x * 256 + threadIdx.x;
    if (tid >= GG * OUTD) return;
    int gi = tid >> 6, oc = tid & 63;
    float s = fcb[oc];
    for (int dd = 0; dd < DD; ++dd) s += pooled[gi * DD + dd] * fcw[dd * OUTD + oc];
    out[tid] = s;
}

/* ===================== host orchestration ===================== */
extern "C" void kernel_launch(void* const* d_in, const int* in_sizes, int n_in,
                              void* d_out, int out_size, void* d_ws, size_t ws_size,
                              hipStream_t stream)
{
    (void)in_sizes; (void)n_in; (void)out_size; (void)ws_size;
    /* jax pytree flatten order: sorted dict keys at every level */
    const int*   p_batch = (const int*)d_in[0];
    const int*   p_ei    = (const int*)d_in[1];
    const float* c_bk = (const float*)d_in[2],  *c_bo = (const float*)d_in[3];
    const float* c_bq = (const float*)d_in[4],  *c_bv = (const float*)d_in[5];
    const float* c_wk = (const float*)d_in[6],  *c_wo = (const float*)d_in[7];
    const float* c_wq = (const float*)d_in[8],  *c_wv = (const float*)d_in[9];
    const float* ff1_b = (const float*)d_in[10], *ff1_w = (const float*)d_in[11];
    const float* ff2_b = (const float*)d_in[12], *ff2_w = (const float*)d_in[13];
    const float* ln1_b = (const float*)d_in[14], *ln1_s = (const float*)d_in[15];
    const float* ln2_b = (const float*)d_in[16], *ln2_s = (const float*)d_in[17];
    const float* ln3_b = (const float*)d_in[18], *ln3_s = (const float*)d_in[19];
    const float* s_bk = (const float*)d_in[20], *s_bo = (const float*)d_in[21];
    const float* s_bq = (const float*)d_in[22], *s_bv = (const float*)d_in[23];
    const float* s_wk = (const float*)d_in[24], *s_wo = (const float*)d_in[25];
    const float* s_wq = (const float*)d_in[26], *s_wv = (const float*)d_in[27];
    const float* fc_b = (const float*)d_in[28], *fc_w = (const float*)d_in[29];
    const float* feat_b = (const float*)d_in[30], *feat_w = (const float*)d_in[31];
    const float* g1_b = (const float*)d_in[32], *g1_w = (const float*)d_in[33];
    const float* g2_b = (const float*)d_in[34], *g2_w = (const float*)d_in[35];
    const float* gat_ad[5], *gat_as[5], *gat_b[5], *gat_w[5];
    gat_ad[0] = (const float*)d_in[36]; gat_as[0] = (const float*)d_in[37];
    gat_b[0]  = (const float*)d_in[38]; gat_w[0]  = (const float*)d_in[39];
    for (int k = 0; k < 4; ++k) {
        gat_ad[k + 1] = (const float*)d_in[40 + 4 * k];
        gat_as[k + 1] = (const float*)d_in[41 + 4 * k];
        gat_b[k + 1]  = (const float*)d_in[42 + 4 * k];
        gat_w[k + 1]  = (const float*)d_in[43 + 4 * k];
    }
    const float* x = (const float*)d_in[56];
    float* outp = (float*)d_out;

    /* workspace layout */
    char* Wb = (char*)d_ws;
    size_t off = 0;
    auto AL = [&](size_t b) { size_t r = off; off += (b + 255) & ~(size_t)255; return r; };
    float* h    = (float*)(Wb + AL((size_t)NN * DD * 4));
    unsigned short* hbf  = (unsigned short*)(Wb + AL((size_t)NN * DD * 2));
    unsigned short* mem  = (unsigned short*)(Wb + AL((size_t)NN * DD * 2));
    float* t2   = (float*)(Wb + AL((size_t)NN * DD * 4));
    unsigned short* t2bf = (unsigned short*)(Wb + AL((size_t)NN * DD * 2));
    float* dlt  = (float*)(Wb + AL((size_t)NN * DD * 4));
    unsigned short* qbf = (unsigned short*)(Wb + AL((size_t)NN * DD * 2));
    unsigned short* kbf = (unsigned short*)(Wb + AL((size_t)NN * DD * 2));
    unsigned short* vbf = (unsigned short*)(Wb + AL((size_t)NN * DD * 2));
    unsigned short* vT  = (unsigned short*)(Wb + AL((size_t)NN * DD * 2));
    unsigned short* ffbf = (unsigned short*)(Wb + AL((size_t)NN * FFH * 2));
    char* bigp = Wb + AL((size_t)NN * FFH * 4);       /* phase-shared region */
    float* hw   = (float*)bigp;                        /* GAT: projection */
    float* elog = (float*)(bigp + (size_t)NN * DD * 4);/* GAT: edge logits */
    float* g1f  = (float*)bigp;                        /* gate: N x FFH f32 */
    float* asrc = (float*)(Wb + AL((size_t)NN * HH * 4));
    float* adst = (float*)(Wb + AL((size_t)NN * HH * 4));
    unsigned* mm = (unsigned*)(Wb + AL((size_t)NN * HH * 4));
    float* den  = (float*)(Wb + AL((size_t)NN * HH * 4));
    float* gbuf = (float*)(Wb + AL((size_t)NN * 4));
    float* gebf = (float*)(Wb + AL((size_t)NN * 4));
    unsigned* gm = (unsigned*)(Wb + AL((size_t)GG * 4));
    float* gs   = (float*)(Wb + AL((size_t)GG * 4));
    float* pooled = (float*)(Wb + AL((size_t)GG * DD * 4));
    unsigned short* xbf = (unsigned short*)(Wb + AL((size_t)NN * FF * 2));
    unsigned short* PK  = (unsigned short*)(Wb + AL((size_t)6373376 * 2));

    auto B256 = [](int n) { return (n + 255) / 256; };
    size_t po = 0;
    auto packw = [&](const float* Wsrc, int K, int Nc) -> unsigned short* {
        unsigned short* dst = PK + po; po += (size_t)K * Nc;
        k_pack_b<<<B256(K * Nc), 256, 0, stream>>>(Wsrc, dst, K, Nc);
        return dst;
    };
    auto gemm = [&](const unsigned short* A, const unsigned short* Bp, const float* bias,
                    float* Cf, unsigned short* Cb, int M, int K, int Nc, int act) {
        dim3 g(Nc / 128, M / 128);
        k_gemm_bf16<<<g, 256, 0, stream>>>(A, Bp, bias, Cf, Cb, M, K, Nc, act);
    };

    /* pack all weights */
    unsigned short* Pfeat = packw(feat_w, FF, EMB);
    unsigned short* Pgat[5];
    Pgat[0] = packw(gat_w[0], EMB, DD);
    for (int k = 1; k < 5; ++k) Pgat[k] = packw(gat_w[k], DD, DD);
    unsigned short* Psq = packw(s_wq, DD, DD), *Psk = packw(s_wk, DD, DD);
    unsigned short* Psv = packw(s_wv, DD, DD), *Pso = packw(s_wo, DD, DD);
    unsigned short* Pcq = packw(c_wq, DD, DD), *Pck = packw(c_wk, DD, DD);
    unsigned short* Pcv = packw(c_wv, DD, DD), *Pco = packw(c_wo, DD, DD);
    unsigned short* Pf1 = packw(ff1_w, DD, FFH), *Pf2 = packw(ff2_w, FFH, DD);
    unsigned short* Pg1 = packw(g1_w, DD, FFH);

    /* feature embedding */
    k_cvt_bf16<<<B256(NN * FF), 256, 0, stream>>>(x, xbf, NN * FF);
    gemm(xbf, Pfeat, feat_b, h, hbf, NN, FF, EMB, 0);

    /* GAT stack */
    for (int L = 0; L < 5; ++L) {
        int din = (L == 0) ? EMB : DD;
        gemm(hbf, Pgat[L], nullptr, hw, nullptr, NN, din, DD, 0);
        k_attn_coef<<<B256(NN * HH), 256, 0, stream>>>(hw, gat_as[L], gat_ad[L], asrc, adst);
        k_fill_u32<<<B256(NN * HH), 256, 0, stream>>>(mm, 0u, NN * HH);
        k_fill_f32<<<B256(NN * HH), 256, 0, stream>>>(den, 0.f, NN * HH);
        k_edge_max<<<B256(EP * HH), 256, 0, stream>>>(p_ei, asrc, adst, elog, mm);
        k_edge_exp<<<B256(EP * HH), 256, 0, stream>>>(p_ei, elog, mm, den);
        k_edge_alpha<<<B256(EP * HH), 256, 0, stream>>>(p_ei, elog, den);
        k_fill_f32<<<B256(NN * DD), 256, 0, stream>>>(t2, 0.f, NN * DD);
        k_scatter<<<B256(EP * DD), 256, 0, stream>>>(p_ei, elog, hw, t2);
        k_bias_leaky<<<B256(NN * DD), 256, 0, stream>>>(t2, gat_b[L], h, hbf);
    }

    /* decoder layer */
    k_copy_u16<<<B256(NN * DD), 256, 0, stream>>>(hbf, mem, NN * DD);
    auto mha = [&](const unsigned short* qin, const unsigned short* kvin,
                   unsigned short* Pq, const float* bq, unsigned short* Pk2, const float* bk,
                   unsigned short* Pv, const float* bv, unsigned short* Po, const float* bo) {
        gemm(qin, Pq, bq, nullptr, qbf, NN, DD, DD, 0);
        gemm(kvin, Pk2, bk, nullptr, kbf, NN, DD, DD, 0);
        gemm(kvin, Pv, bv, nullptr, vbf, NN, DD, DD, 0);
        k_transpose_u16<<<B256(NN * DD), 256, 0, stream>>>(vbf, vT);
        k_flash<<<256, 256, 0, stream>>>(qbf, kbf, vT, t2, t2bf);
        gemm(t2bf, Po, bo, dlt, nullptr, NN, DD, DD, 0);
    };
    mha(hbf, hbf, Psq, s_bq, Psk, s_bk, Psv, s_bv, Pso, s_bo);
    k_ln_res<<<NN / 8, 256, 0, stream>>>(h, dlt, ln1_s, ln1_b, h, hbf);
    mha(hbf, mem, Pcq, c_bq, Pck, c_bk, Pcv, c_bv, Pco, c_bo);
    k_ln_res<<<NN / 8, 256, 0, stream>>>(h, dlt, ln2_s, ln2_b, h, hbf);
    gemm(hbf, Pf1, ff1_b, nullptr, ffbf, NN, DD, FFH, 1);
    gemm(ffbf, Pf2, ff2_b, dlt, nullptr, NN, FFH, DD, 0);
    k_ln_res<<<NN / 8, 256, 0, stream>>>(h, dlt, ln3_s, ln3_b, h, hbf);

    /* gated softmax pooling */
    gemm(hbf, Pg1, g1_b, g1f, nullptr, NN, DD, FFH, 1);
    k_rowdot<<<NN / 8, 256, 0, stream>>>(g1f, g2_w, g2_b, gbuf, FFH);
    k_fill_u32<<<1, 256, 0, stream>>>(gm, 0u, GG);
    k_fill_f32<<<1, 256, 0, stream>>>(gs, 0.f, GG);
    k_gmax<<<B256(NN), 256, 0, stream>>>(p_batch, gbuf, gm);
    k_gexp<<<B256(NN), 256, 0, stream>>>(p_batch, gbuf, gm, gebf, gs);
    k_fill_f32<<<B256(GG * DD), 256, 0, stream>>>(pooled, 0.f, GG * DD);
    k_pool<<<B256(NN * DD), 256, 0, stream>>>(p_batch, gebf, gs, h, pooled);
    k_fc<<<B256(GG * OUTD), 256, 0, stream>>>(pooled, fc_w, fc_b, outp);
}
